// ALiBiMultiHeadAttention_84198538870954
// MI455X (gfx1250) — matrix operand order repaired and checked by reading, not compiled, not run on hardware
//
#include <hip/hip_runtime.h>

typedef _Float16 half_t;
typedef __attribute__((ext_vector_type(16))) _Float16 v16h;
typedef __attribute__((ext_vector_type(8)))  _Float16 v8h;
typedef __attribute__((ext_vector_type(8)))  float    v8f;
typedef __attribute__((ext_vector_type(4)))  unsigned int u32x4;
typedef __attribute__((ext_vector_type(8)))  int          i32x8;
typedef __attribute__((ext_vector_type(4)))  int          i32x4;

#define WMMA_F16(a, b, c) \
  __builtin_amdgcn_wmma_f32_16x16x32_f16(false, (a), false, (b), (short)0, (c), false, false)

#if __has_builtin(__builtin_amdgcn_tensor_load_to_lds) && __has_builtin(__builtin_amdgcn_s_wait_tensorcnt)
#define USE_TDM 1
#else
#define USE_TDM 0
#endif

// Problem constants
static constexpr int TQ = 1024;    // queries per batch
static constexpr int DMODEL = 1024;
static constexpr int DHEAD = 64;
static constexpr int TKTOT = 2048; // cache (1024) + new (1024)
static constexpr int MROWS = 4096; // B*Tq

static __device__ __forceinline__ v16h ld16(const half_t* p0, const half_t* p1) {
  v8h lo = *reinterpret_cast<const v8h*>(p0);
  v8h hi = *reinterpret_cast<const v8h*>(p1);
  v16h r;
#pragma unroll
  for (int i = 0; i < 8; ++i) { r[i] = lo[i]; r[i + 8] = hi[i]; }
  return r;
}

#if USE_TDM
// Issue a TDM 2D tile load (f16 elements) into LDS.
// D# bitfields per CDNA5 ISA (group0) / §8.4 (group1).
static __device__ __forceinline__ void tdm_load_2d(unsigned long long gaddr,  // global byte addr of tile
                                                   unsigned int lds_off,      // LDS byte offset
                                                   unsigned int tensor_d0,    // dim0 length (elems)
                                                   unsigned int tensor_d1,    // dim1 length (elems)
                                                   unsigned long long d0_stride, // dim0 stride (elems)
                                                   unsigned int tile_d0,      // tile dim0 (elems, contiguous)
                                                   unsigned int tile_d1) {    // tile dim1 (rows)
  u32x4 g0;
  g0[0] = (unsigned)__builtin_amdgcn_readfirstlane(1);                        // count=1, no gather
  g0[1] = (unsigned)__builtin_amdgcn_readfirstlane((int)lds_off);             // lds_addr [63:32]
  g0[2] = (unsigned)__builtin_amdgcn_readfirstlane((int)(unsigned)gaddr);     // global_addr [95:64]
  g0[3] = (unsigned)__builtin_amdgcn_readfirstlane(
            (int)(((unsigned)(gaddr >> 32) & 0x1FFFFFFu) | (2u << 30)));      // addr[56:32] | type=2
  i32x8 g1;
  g1[0] = __builtin_amdgcn_readfirstlane((int)(1u << 16));                    // wg_mask=0, data_size=2B
  g1[1] = __builtin_amdgcn_readfirstlane((int)((tensor_d0 & 0xFFFFu) << 16)); // atomic_addr=0 | dim0 lo16
  g1[2] = __builtin_amdgcn_readfirstlane((int)((tensor_d0 >> 16) | ((tensor_d1 & 0xFFFFu) << 16)));
  g1[3] = __builtin_amdgcn_readfirstlane((int)((tensor_d1 >> 16) | (tile_d0 << 16)));
  g1[4] = __builtin_amdgcn_readfirstlane((int)(tile_d1 & 0xFFFFu));           // tile_dim1 | tile_dim2=0
  g1[5] = __builtin_amdgcn_readfirstlane((int)(unsigned)d0_stride);           // dim0_stride lo32
  g1[6] = __builtin_amdgcn_readfirstlane((int)((unsigned)(d0_stride >> 32) & 0xFFFFu));
  g1[7] = 0;                                                                  // dim1_stride (unused, 2D)
  i32x4 z4 = {0, 0, 0, 0};
#if defined(__clang_major__) && (__clang_major__ >= 23)
  i32x8 z8 = {0, 0, 0, 0, 0, 0, 0, 0};
  __builtin_amdgcn_tensor_load_to_lds(g0, g1, z4, z4, z8, 0);
#else
  __builtin_amdgcn_tensor_load_to_lds(g0, g1, z4, z4, 0);
#endif
}
#endif

// ---------------- prep kernels ----------------
__global__ void k_f32_to_f16(const float* __restrict__ in, half_t* __restrict__ out, int n) {
  int i = blockIdx.x * blockDim.x + threadIdx.x;
  if (i < n) out[i] = (half_t)in[i];
}

// W [K=1024][N=1024] fp32 row-major -> Wt [N][K] f16 (column of B contiguous)
__global__ void k_transpose_w(const float* __restrict__ W, half_t* __restrict__ Wt) {
  int i = blockIdx.x * blockDim.x + threadIdx.x; // i = k*1024 + n
  int k = i >> 10, n = i & 1023;
  Wt[(size_t)n * 1024 + k] = (half_t)W[i];
}

// k_cache fp32 (B,1024,1024) -> kb f16 at [b][t][d], t in [0,1024)
__global__ void k_pack_kcache(const float* __restrict__ in, half_t* __restrict__ kb) {
  int i = blockIdx.x * blockDim.x + threadIdx.x;
  int b = i >> 20;
  int rest = i & ((1 << 20) - 1);
  kb[(size_t)b * (TKTOT * DMODEL) + rest] = (half_t)in[i];
}

// v_cache fp32 (B,1024(t),1024(d)) -> vT f16 [b][d][t], t in [0,1024)
__global__ void k_pack_vcache(const float* __restrict__ in, half_t* __restrict__ vT) {
  int i = blockIdx.x * blockDim.x + threadIdx.x;
  int b = i >> 20;
  int t = (i >> 10) & 1023;
  int d = i & 1023;
  vT[(size_t)b * (DMODEL * TKTOT) + (size_t)d * TKTOT + t] = (half_t)in[i];
}

// ---------------- WMMA GEMM: C[4096][1024] = A[M][K=1024] * Bt[N][K]^T (+bias) ----------------
// mode 0: q -> outB f16 [m][n]   mode 1: k -> kb[b][1024+t][n] + fp32 k_cache_new
// mode 2: v -> vT[b][n][1024+t] + fp32 v_cache_new            mode 3: o -> fp32 out
__launch_bounds__(256)
__global__ void k_gemm(const half_t* __restrict__ A, const half_t* __restrict__ Bt,
                       const float* __restrict__ bias, float* __restrict__ outF,
                       half_t* __restrict__ outB, int mode) {
  const int lane = threadIdx.x & 31;
  const int w    = threadIdx.x >> 5;
  const int g    = lane >> 4;
  const int ln   = lane & 15;
  const int m0   = blockIdx.x * 64 + (w >> 2) * 32;   // wave tile: 32 x 64
  const int n0   = blockIdx.y * 256 + (w & 3) * 64;
  const int K    = 1024;

  v8f acc[2][4] = {};

  for (int k0 = 0; k0 < K; k0 += 32) {
    v16h af[2], bf[4];
#pragma unroll
    for (int mt = 0; mt < 2; ++mt) {
      const half_t* p = A + (size_t)(m0 + mt * 16 + ln) * K + k0 + 16 * g;
      af[mt] = ld16(p, p + 8);                        // A-frag: K = 16g+e contiguous
    }
#pragma unroll
    for (int nt = 0; nt < 4; ++nt) {
      const half_t* p = Bt + (size_t)(n0 + nt * 16 + ln) * K + k0 + 16 * g;
      bf[nt] = *reinterpret_cast<const v16h*>(p);     // B-frag: K = 16g+e contiguous
    }
#pragma unroll
    for (int mt = 0; mt < 2; ++mt)
#pragma unroll
      for (int nt = 0; nt < 4; ++nt)
        acc[mt][nt] = WMMA_F16(af[mt], bf[nt], acc[mt][nt]);
  }

#pragma unroll
  for (int mt = 0; mt < 2; ++mt)
#pragma unroll
    for (int nt = 0; nt < 4; ++nt)
#pragma unroll
      for (int r = 0; r < 8; ++r) {
        int m = m0 + mt * 16 + r + 8 * g;
        int n = n0 + nt * 16 + ln;
        float v = acc[mt][nt][r];
        if (bias) v += bias[n];
        if (mode == 0) {
          outB[(size_t)m * DMODEL + n] = (half_t)v;
        } else if (mode == 1) {
          int bb = m >> 10, t = m & 1023;
          outB[(size_t)bb * (TKTOT * DMODEL) + (size_t)(1024 + t) * DMODEL + n] = (half_t)v;
          outF[(size_t)m * DMODEL + n] = v;
        } else if (mode == 2) {
          int bb = m >> 10, t = m & 1023;
          outB[(size_t)bb * (DMODEL * TKTOT) + (size_t)n * TKTOT + 1024 + t] = (half_t)v;
          outF[(size_t)m * DMODEL + n] = v;
        } else {
          outF[(size_t)m * DMODEL + n] = v;
        }
      }
}

// ---------------- flash attention with ALiBi: TDM double-buffered K/V staging + WMMA ----------------
__launch_bounds__(256)
__global__ void k_attn(const half_t* __restrict__ qb, const half_t* __restrict__ kb,
                       const half_t* __restrict__ vT, const float* __restrict__ mask,
                       half_t* __restrict__ ab) {
  __shared__ __align__(16) half_t plds[8][16][32];      // per-wave P re-striping tile
#if USE_TDM
  __shared__ __align__(16) half_t ktile[8][2][32 * 64]; // per-wave double-buffered K tile [key][dh]
  __shared__ __align__(16) half_t vtile[8][2][64 * 32]; // per-wave double-buffered V tile [dh][key]
#endif
  const int lane = threadIdx.x & 31;
  const int w    = threadIdx.x >> 5;
  const int g    = lane >> 4;
  const int ln   = lane & 15;
  const int gw   = blockIdx.x * 8 + w;       // 4096 wave tiles
  const int qt   = gw & 63;
  const int h    = (gw >> 6) & 15;
  const int b    = gw >> 10;
  const int q0   = qt * 16;
  const float slope = exp2f(-0.5f * (float)(h + 1));
  const float scale = 0.125f;                // 1/sqrt(64)

  // Q A-fragments (loop invariant)
  const half_t* qrow = qb + (size_t)(b * TQ + q0 + ln) * DMODEL + h * DHEAD;
  v16h qa0 = ld16(qrow + 16 * g,      qrow + 8 + 16 * g);
  v16h qa1 = ld16(qrow + 32 + 16 * g, qrow + 40 + 16 * g);

  const half_t* kbase = kb + (size_t)b * (TKTOT * DMODEL) + h * DHEAD;
  const half_t* vbase = vT + (size_t)b * (DMODEL * TKTOT) + (size_t)(h * DHEAD) * TKTOT;

  v8f O[4] = {};
  float mr[8], lr[8];
#pragma unroll
  for (int r = 0; r < 8; ++r) { mr[r] = -3.0e38f; lr[r] = 0.f; }

#if USE_TDM
  // preload chunk 0 into buffer 0: K tile (tile 64 x 32 rows) and V tile (tile 32 x 64 rows)
  tdm_load_2d((unsigned long long)(uintptr_t)kbase * 1ull + 0ull,
              (unsigned int)(uintptr_t)&ktile[w][0][0], DMODEL, TKTOT, DMODEL, DHEAD, 32);
  tdm_load_2d((unsigned long long)(uintptr_t)vbase,
              (unsigned int)(uintptr_t)&vtile[w][0][0], TKTOT, DHEAD, TKTOT, 32, DHEAD);
#endif

  int buf = 0;
  for (int kc = 0; kc < TKTOT; kc += 32, buf ^= 1) {
#if USE_TDM
    if (kc + 32 < TKTOT) {  // issue next chunk's DMA, then wait for current (in-order TENSORcnt)
      tdm_load_2d((unsigned long long)(uintptr_t)(kbase + (size_t)(kc + 32) * DMODEL),
                  (unsigned int)(uintptr_t)&ktile[w][buf ^ 1][0], DMODEL, TKTOT, DMODEL, DHEAD, 32);
      tdm_load_2d((unsigned long long)(uintptr_t)(vbase + (kc + 32)),
                  (unsigned int)(uintptr_t)&vtile[w][buf ^ 1][0], TKTOT, DHEAD, TKTOT, 32, DHEAD);
      __builtin_amdgcn_s_wait_tensorcnt((short)2);
    } else {
      __builtin_amdgcn_s_wait_tensorcnt((short)0);
    }
    const half_t* kt = &ktile[w][buf][0];
    const half_t* vt = &vtile[w][buf][0];
    v16h kb0a = *reinterpret_cast<const v16h*>(kt + (size_t)ln * DHEAD + 16 * g);
    v16h kb1a = *reinterpret_cast<const v16h*>(kt + (size_t)ln * DHEAD + 32 + 16 * g);
    v16h kb0b = *reinterpret_cast<const v16h*>(kt + (size_t)(16 + ln) * DHEAD + 16 * g);
    v16h kb1b = *reinterpret_cast<const v16h*>(kt + (size_t)(16 + ln) * DHEAD + 32 + 16 * g);
#else
    const half_t* kp0 = kbase + (size_t)(kc + ln) * DMODEL + 16 * g;
    const half_t* kp1 = kbase + (size_t)(kc + 16 + ln) * DMODEL + 16 * g;
    v16h kb0a = *reinterpret_cast<const v16h*>(kp0);
    v16h kb1a = *reinterpret_cast<const v16h*>(kp0 + 32);
    v16h kb0b = *reinterpret_cast<const v16h*>(kp1);
    v16h kb1b = *reinterpret_cast<const v16h*>(kp1 + 32);
#endif
    v8f S0 = {}, S1 = {};
    S0 = WMMA_F16(qa0, kb0a, S0);
    S0 = WMMA_F16(qa1, kb1a, S0);
    S1 = WMMA_F16(qa0, kb0b, S1);
    S1 = WMMA_F16(qa1, kb1b, S1);

    const int key0 = kc + ln, key1 = kc + 16 + ln;
    const float bias0 = -slope * (float)(TKTOT - 1 - key0);
    const float bias1 = -slope * (float)(TKTOT - 1 - key1);

    float p0[8], p1[8], alpha[8];
#pragma unroll
    for (int r = 0; r < 8; ++r) {
      int qg = q0 + r + 8 * g;
      float s0 = S0[r] * scale + mask[(size_t)qg * TKTOT + key0] + bias0;
      float s1 = S1[r] * scale + mask[(size_t)qg * TKTOT + key1] + bias1;
      float t = fmaxf(s0, s1);
      t = fmaxf(t, __shfl_xor(t, 1));
      t = fmaxf(t, __shfl_xor(t, 2));
      t = fmaxf(t, __shfl_xor(t, 4));
      t = fmaxf(t, __shfl_xor(t, 8));
      float mnew = fmaxf(mr[r], t);
      alpha[r] = __expf(mr[r] - mnew);
      mr[r] = mnew;
      p0[r] = __expf(s0 - mnew);
      p1[r] = __expf(s1 - mnew);
      float rs = p0[r] + p1[r];
      rs += __shfl_xor(rs, 1);
      rs += __shfl_xor(rs, 2);
      rs += __shfl_xor(rs, 4);
      rs += __shfl_xor(rs, 8);
      lr[r] = lr[r] * alpha[r] + rs;
    }
#pragma unroll
    for (int n = 0; n < 4; ++n)
#pragma unroll
      for (int r = 0; r < 8; ++r) O[n][r] *= alpha[r];

    // Re-stripe P (C-layout) into A-layout through per-wave LDS tile
#pragma unroll
    for (int r = 0; r < 8; ++r) {
      plds[w][r + 8 * g][ln]      = (half_t)p0[r];
      plds[w][r + 8 * g][16 + ln] = (half_t)p1[r];
    }
    asm volatile("s_wait_dscnt 0" ::: "memory");
    const half_t* pp = &plds[w][ln][0];
    v16h Pa = ld16(pp + 16 * g, pp + 8 + 16 * g);

#pragma unroll
    for (int n = 0; n < 4; ++n) {
#if USE_TDM
      v16h vf = *reinterpret_cast<const v16h*>(vt + (size_t)(16 * n + ln) * 32 + 16 * g);
#else
      v16h vf = *reinterpret_cast<const v16h*>(vbase + (size_t)(16 * n + ln) * TKTOT + kc + 16 * g);
#endif
      O[n] = WMMA_F16(Pa, vf, O[n]);
    }
  }

#pragma unroll
  for (int r = 0; r < 8; ++r) {
    float inv = 1.0f / lr[r];
    int m = b * TQ + q0 + r + 8 * g;
#pragma unroll
    for (int n = 0; n < 4; ++n)
      ab[(size_t)m * DMODEL + h * DHEAD + 16 * n + ln] = (half_t)(O[n][r] * inv);
  }
}

// ---------------- host launch ----------------
extern "C" void kernel_launch(void* const* d_in, const int* in_sizes, int n_in,
                              void* d_out, int out_size, void* d_ws, size_t ws_size,
                              hipStream_t stream) {
  const float* x    = (const float*)d_in[0];
  const float* kcin = (const float*)d_in[1];
  const float* vcin = (const float*)d_in[2];
  const float* mask = (const float*)d_in[3];
  const float* Wq   = (const float*)d_in[4];
  const float* bq   = (const float*)d_in[5];
  const float* Wk   = (const float*)d_in[6];
  const float* Wv   = (const float*)d_in[7];
  const float* bv   = (const float*)d_in[8];
  const float* Wo   = (const float*)d_in[9];
  const float* bo   = (const float*)d_in[10];

  char* ws = (char*)d_ws;
  half_t* xb  = (half_t*)(ws);                         // 8 MB (reused as ab)
  half_t* qb  = (half_t*)(ws + ((size_t)8  << 20));    // 8 MB
  half_t* WtQ = (half_t*)(ws + ((size_t)16 << 20));    // 2 MB each
  half_t* WtK = (half_t*)(ws + ((size_t)18 << 20));
  half_t* WtV = (half_t*)(ws + ((size_t)20 << 20));
  half_t* WtO = (half_t*)(ws + ((size_t)22 << 20));
  half_t* kbf = (half_t*)(ws + ((size_t)24 << 20));    // 16 MB: [b][2048][1024]
  half_t* vTb = (half_t*)(ws + ((size_t)40 << 20));    // 16 MB: [b][1024][2048]
  half_t* ab  = xb;                                    // reuse x-f16 region

  float* out  = (float*)d_out;                         // (B,Tq,D)
  float* kOut = out + (size_t)4 * 1024 * 1024;         // k_cache_new
  float* vOut = out + (size_t)8 * 1024 * 1024;         // v_cache_new

  const int TPB = 256;
  const int NX = MROWS * DMODEL;                       // 4M elems

  k_f32_to_f16 <<<NX / TPB, TPB, 0, stream>>>(x, xb, NX);
  k_transpose_w<<<(DMODEL * DMODEL) / TPB, TPB, 0, stream>>>(Wq, WtQ);
  k_transpose_w<<<(DMODEL * DMODEL) / TPB, TPB, 0, stream>>>(Wk, WtK);
  k_transpose_w<<<(DMODEL * DMODEL) / TPB, TPB, 0, stream>>>(Wv, WtV);
  k_transpose_w<<<(DMODEL * DMODEL) / TPB, TPB, 0, stream>>>(Wo, WtO);
  k_pack_kcache<<<NX / TPB, TPB, 0, stream>>>(kcin, kbf);
  k_pack_vcache<<<NX / TPB, TPB, 0, stream>>>(vcin, vTb);

  dim3 ggrid(MROWS / 64, DMODEL / 256);
  k_gemm<<<ggrid, TPB, 0, stream>>>(xb, WtQ, bq, nullptr, qb, 0);
  k_gemm<<<ggrid, TPB, 0, stream>>>(xb, WtK, nullptr, kOut, kbf, 1);
  k_gemm<<<ggrid, TPB, 0, stream>>>(xb, WtV, bv, vOut, vTb, 2);

  k_attn<<<512, TPB, 0, stream>>>(qb, kbf, vTb, mask, ab);

  k_gemm<<<ggrid, TPB, 0, stream>>>(ab, WtO, bo, out, nullptr, 3);
}
